// DCNMLayer_6511170420805
// MI455X (gfx1250) — matrix operand order repaired and checked by reading, not compiled, not run on hardware
//
#include <hip/hip_runtime.h>
#include <hip/hip_bf16.h>
#include <math.h>

// ---------------------------------------------------------------------------
// DCN-V2 mixture-of-low-rank-experts, fused for gfx1250 (MI455X).
//   per layer:  [v | gate_logits] = x_l @ [Vcat | gateW_pad]   (one WMMA pass,
//                                          N = 256 + 16, K = 1024)
//               g = softmax(logits)                       (VALU, 4-wide)
//               c = tanh(C_e @ v_e) * g_e                 (WMMA bf16, K=64)
//               Z = c @ Ucat                              (WMMA bf16, K=256)
//               x_l = x_l + x0 * (Z + bias)
// Both layers fused per 64-row tile; weights pre-converted to bf16 N-major.
// ---------------------------------------------------------------------------

typedef __bf16 bf16_t;
typedef __attribute__((ext_vector_type(8)))  __bf16 v8bf;
typedef __attribute__((ext_vector_type(16))) __bf16 v16bf;
typedef __attribute__((ext_vector_type(8)))  float  v8f;

constexpr int Dd  = 1024;   // feature dim
constexpr int Rr  = 64;     // rank
constexpr int Ee  = 4;      // experts
constexpr int Ll  = 2;      // layers
constexpr int NER = Ee * Rr;        // 256 = fused expert*rank dim
constexpr int NV  = NER + 16;       // 272 = GEMM1 N (256 v-cols + 16 gate pad)

constexpr int MT      = 64;             // rows per workgroup
constexpr int WAVES   = MT / 16;        // 4 waves, one 16-row strip each
constexpr int THREADS = WAVES * 32;     // 128 (wave32)
constexpr int XS = Dd  + 16;            // LDS stride (bf16) — 16B aligned, conflict-free
constexpr int VS = NER + 16;

// ---- fast tanh: gfx1250 has hardware v_tanh_f32 ---------------------------
__device__ __forceinline__ float fast_tanh(float x) {
#if defined(__AMDGCN__) && __has_builtin(__builtin_amdgcn_tanhf)
  return __builtin_amdgcn_tanhf(x);
#else
  float e = __expf(2.0f * fabsf(x));
  return copysignf(1.0f - 2.0f / (e + 1.0f), x);
#endif
}

// ---- WMMA fragment helpers (layouts per CDNA5 ISA 7.12.2) -----------------
// A 16x32 bf16: lane<16 -> K {k0..k0+7, k0+16..k0+23}; lane>=16 -> +8 offset.
__device__ __forceinline__ v16bf frag_split(const bf16_t* p) {
  v8bf lo = *reinterpret_cast<const v8bf*>(p);
  v8bf hi = *reinterpret_cast<const v8bf*>(p + 16);
  return __builtin_shufflevector(lo, hi, 0,1,2,3,4,5,6,7,8,9,10,11,12,13,14,15);
}
// B 32x16 bf16 from N-major ([N][K]) memory: lane n holds B[{k..k+7, k+16..k+23}, n].
__device__ __forceinline__ v16bf frag_contig(const bf16_t* p) {
  v8bf lo = *reinterpret_cast<const v8bf*>(p);
  v8bf hi = *reinterpret_cast<const v8bf*>(p + 16);
  return __builtin_shufflevector(lo, hi, 0,1,2,3,4,5,6,7,8,9,10,11,12,13,14,15);
}
__device__ __forceinline__ v8f wmma_bf16(v16bf a, v16bf b, v8f c) {
  return __builtin_amdgcn_wmma_f32_16x16x32_bf16(false, a, false, b, (short)0, c,
                                                 false, false);
}

// ---- weight prep: fp32 -> bf16, N-major -----------------------------------
// Vg[l][n][d]: n<256 -> V[l][e][d][r] (n=e*R+r); 256..259 -> gateW[n-256][d]; else 0
__global__ void prep_Vg(const float* __restrict__ V, const float* __restrict__ gateW,
                        bf16_t* __restrict__ Vg) {
  for (int idx = blockIdx.x * blockDim.x + threadIdx.x; idx < Ll * NV * Dd;
       idx += gridDim.x * blockDim.x) {
    int l = idx / (NV * Dd), rem = idx % (NV * Dd);
    int n = rem / Dd, d = rem % Dd;
    float val;
    if (n < NER) {
      int e = n >> 6, r = n & 63;
      val = V[(((size_t)l * Ee + e) * Dd + d) * Rr + r];
    } else if (n < NER + Ee) {
      val = gateW[(size_t)(n - NER) * Dd + d];
    } else {
      val = 0.0f;
    }
    Vg[idx] = (bf16_t)val;
  }
}
// Ut[l][d][e*R+r] = U[l][e][d][r]      (B^T for GEMM2: N=1024 rows, K=256)
__global__ void prep_Ut(const float* __restrict__ U, bf16_t* __restrict__ Ut) {
  for (int idx = blockIdx.x * blockDim.x + threadIdx.x; idx < Ll * Dd * NER;
       idx += gridDim.x * blockDim.x) {
    int l = idx / (Dd * NER), rem = idx % (Dd * NER);
    int d = rem / NER, er = rem % NER;
    int e = er >> 6, r = er & 63;
    Ut[idx] = (bf16_t)U[(((size_t)l * Ee + e) * Dd + d) * Rr + r];
  }
}
// Cb[l][e][r][q] = C[l][e][r][q]       (already B^T layout: N=r rows, K=q)
__global__ void prep_Cb(const float* __restrict__ C, bf16_t* __restrict__ Cb, int n) {
  for (int idx = blockIdx.x * blockDim.x + threadIdx.x; idx < n;
       idx += gridDim.x * blockDim.x)
    Cb[idx] = (bf16_t)C[idx];
}

// ---- fused cross-network kernel -------------------------------------------
__global__ __launch_bounds__(THREADS, 1)
void dcn_moe_fused(const float*  __restrict__ x0g,
                   const bf16_t* __restrict__ Vg,
                   const bf16_t* __restrict__ Ut,
                   const bf16_t* __restrict__ Cb,
                   const float*  __restrict__ bias,
                   float*        __restrict__ out) {
  extern __shared__ char smem_raw[];
  bf16_t* xl   = (bf16_t*)smem_raw;          // [MT][XS]  current layer input (bf16)
  bf16_t* vb   = xl + MT * XS;               // [MT][VS]  tanh(V^T x)
  bf16_t* wb   = vb + MT * VS;               // [MT][VS]  gate-scaled tanh(C v)
  float*  gate = (float*)(wb + MT * VS);     // [MT][Ee]  softmax weights

  const int tid  = threadIdx.x;
  const int wave = tid >> 5;
  const int lane = tid & 31;
  const int lan  = lane & 15;
  const int rowBase = blockIdx.x * MT;

  const int mrow = wave * 16 + lan;          // A-fragment source row (LDS)
  const int aK   = (lane < 16) ? 0 : 8;      // A split offset
  const int bK   = (lane < 16) ? 0 : 8;      // B split offset
  const int m0   = (lane < 16) ? 0 : 8;      // D-fragment row base

  // stage x0 tile into LDS as bf16, vectorized (2x float4 -> 1x v8bf per step)
  for (int idx = tid; idx < MT * (Dd / 8); idx += THREADS) {
    int r = idx >> 7, c8 = (idx & 127) * 8;
    const float4* gp =
        reinterpret_cast<const float4*>(&x0g[(size_t)(rowBase + r) * Dd + c8]);
    float4 f0 = gp[0], f1 = gp[1];
    v8bf h = {(bf16_t)f0.x, (bf16_t)f0.y, (bf16_t)f0.z, (bf16_t)f0.w,
              (bf16_t)f1.x, (bf16_t)f1.y, (bf16_t)f1.z, (bf16_t)f1.w};
    *reinterpret_cast<v8bf*>(&xl[r * XS + c8]) = h;
  }
  __syncthreads();
  // After this point every phase touches only this wave's 16-row strip, so the
  // layer body needs no cross-wave barriers (one per layer for cache pacing).

  for (int l = 0; l < Ll; ++l) {
    // ---- GEMM1 (+gate): [v | logits] = x_l @ Vg   M=MT, N=272, K=1024 ----
    {
      v8f acc[17] = {};
      const bf16_t* bprow = Vg + (size_t)l * NV * Dd + (size_t)lan * Dd + bK;
      for (int ks = 0; ks < Dd / 32; ++ks) {
        const int k0 = ks * 32;
        const v16bf a = frag_split(&xl[mrow * XS + k0 + aK]);
        __builtin_prefetch(bprow + k0 + 32, 0, 0);
        #pragma unroll
        for (int nt = 0; nt < 17; ++nt)
          acc[nt] = wmma_bf16(a, frag_contig(bprow + (size_t)nt * 16 * Dd + k0),
                              acc[nt]);
      }
      #pragma unroll
      for (int nt = 0; nt < 16; ++nt)
        #pragma unroll
        for (int i = 0; i < 8; ++i)
          vb[(wave * 16 + m0 + i) * VS + nt * 16 + lan] =
              (bf16_t)fast_tanh(acc[nt][i]);
      // gate logits live in acc[16] (cols 4..15 of that tile are zero weights)
      #pragma unroll
      for (int i = 0; i < 8; ++i)
        if (lan < Ee) gate[(wave * 16 + m0 + i) * Ee + lan] = acc[16][i];
      if (lane < 16) {   // softmax over 4 experts, own-wave rows only
        int r = wave * 16 + lane;
        float g0 = gate[r*4+0], g1 = gate[r*4+1], g2 = gate[r*4+2], g3 = gate[r*4+3];
        float m = fmaxf(fmaxf(g0, g1), fmaxf(g2, g3));
        float e0 = __expf(g0 - m), e1 = __expf(g1 - m);
        float e2 = __expf(g2 - m), e3 = __expf(g3 - m);
        float inv = 1.f / (e0 + e1 + e2 + e3);
        gate[r*4+0] = e0*inv; gate[r*4+1] = e1*inv;
        gate[r*4+2] = e2*inv; gate[r*4+3] = e3*inv;
      }
    }

    // ---- C-mix per expert: w_e = tanh(v_e @ C_e^T) * gate_e   (K=64) ----
    #pragma unroll
    for (int e = 0; e < Ee; ++e) {
      v8f acc[4] = {};
      const bf16_t* Ce = Cb + (size_t)(l * Ee + e) * Rr * Rr;  // [64][64] N-major
      #pragma unroll
      for (int ks = 0; ks < Rr / 32; ++ks) {
        const int k0 = ks * 32;
        const v16bf a = frag_split(&vb[mrow * VS + e * Rr + k0 + aK]);
        const bf16_t* bp = Ce + (size_t)lan * Rr + k0 + bK;
        #pragma unroll
        for (int nt = 0; nt < 4; ++nt)
          acc[nt] = wmma_bf16(a, frag_contig(bp + (size_t)nt * 16 * Rr), acc[nt]);
      }
      #pragma unroll
      for (int nt = 0; nt < 4; ++nt)
        #pragma unroll
        for (int i = 0; i < 8; ++i) {
          int r = wave * 16 + m0 + i;
          wb[r * VS + e * Rr + nt * 16 + lan] =
              (bf16_t)(fast_tanh(acc[nt][i]) * gate[r * Ee + e]);
        }
    }

    // ---- GEMM2 + residual: Z = w @ Ucat ; x_l += x0*(Z+bias)  M=MT,N=1024,K=256
    {
      v16bf aw[8];                      // hoist all K fragments of w (64 VGPRs)
      #pragma unroll
      for (int ks = 0; ks < NER / 32; ++ks)
        aw[ks] = frag_split(&wb[mrow * VS + ks * 32 + aK]);

      const bf16_t* Bt2   = Ut + (size_t)l * Dd * NER;  // [1024][256] N-major
      const float*  biasl = bias + (size_t)l * Dd;
      for (int nc = 0; nc < Dd / 128; ++nc) {           // 8 chunks of 8 N-tiles
        v8f acc[8] = {};
        const bf16_t* bprow = Bt2 + (size_t)(nc * 128 + lan) * NER + bK;
        __builtin_prefetch(bprow + 128 * NER, 0, 0);    // next chunk's row
        #pragma unroll
        for (int ks = 0; ks < NER / 32; ++ks) {
          #pragma unroll
          for (int nt = 0; nt < 8; ++nt)
            acc[nt] = wmma_bf16(aw[ks],
                                frag_contig(bprow + (size_t)nt * 16 * NER + ks * 32),
                                acc[nt]);
        }
        #pragma unroll
        for (int nt = 0; nt < 8; ++nt)
          #pragma unroll
          for (int i = 0; i < 8; ++i) {
            int r    = wave * 16 + m0 + i;
            int dcol = nc * 128 + nt * 16 + lan;
            float xo   = x0g[(size_t)(rowBase + r) * Dd + dcol];
            float xnew = (float)xl[r * XS + dcol] + xo * (acc[nt][i] + biasl[dcol]);
            if (l == Ll - 1)
              out[(size_t)(rowBase + r) * Dd + dcol] = xnew;
            else
              xl[r * XS + dcol] = (bf16_t)xnew;   // own-wave rows only: race-free
          }
      }
    }
    __syncthreads();   // pace waves so they share B streams in the WGP$
  }
}

// ---------------------------------------------------------------------------
extern "C" void kernel_launch(void* const* d_in, const int* in_sizes, int n_in,
                              void* d_out, int out_size, void* d_ws, size_t ws_size,
                              hipStream_t stream) {
  const float* x     = (const float*)d_in[0];
  const float* U     = (const float*)d_in[1];
  const float* V     = (const float*)d_in[2];
  const float* C     = (const float*)d_in[3];
  const float* bias  = (const float*)d_in[4];
  const float* gateW = (const float*)d_in[5];
  float*       out   = (float*)d_out;

  const int Brows = in_sizes[0] / Dd;   // 16384

  // workspace: bf16 N-major weights (~2.3 MB total)
  bf16_t* Vg = (bf16_t*)d_ws;                  // L*272*1024  (Vcat + padded gateW)
  bf16_t* Ut = Vg + (size_t)Ll * NV * Dd;      // L*1024*256
  bf16_t* Cb = Ut + (size_t)Ll * Dd * NER;     // L*E*64*64

  { int n = Ll * NV * Dd;      prep_Vg<<<(n + 255) / 256, 256, 0, stream>>>(V, gateW, Vg); }
  { int n = Ll * Dd * NER;     prep_Ut<<<(n + 255) / 256, 256, 0, stream>>>(U, Ut); }
  { int n = Ll * Ee * Rr * Rr; prep_Cb<<<(n + 255) / 256, 256, 0, stream>>>(C, Cb, n); }

  const size_t shmem = (size_t)MT * XS * sizeof(bf16_t)
                     + 2 * (size_t)MT * VS * sizeof(bf16_t)
                     + (size_t)MT * Ee * sizeof(float);   // ~204 KB (< 320 KB WGP LDS)

  dcn_moe_fused<<<Brows / MT, THREADS, shmem, stream>>>(x, Vg, Ut, Cb, bias, out);
}
